// KeypointDetector_70136815943702
// MI455X (gfx1250) — matrix operand order repaired and checked by reading, not compiled, not run on hardware
//
#include <hip/hip_runtime.h>
#include <hip/hip_bf16.h>
#include <math.h>

// ---------------------------------------------------------------------------
// KeypointDetector for MI455X (gfx1250, wave32, WMMA).
//  - All WMMA operands pre-swizzled into WMMA lane layout ("fragment format", contiguous k = 16*hi + h):
//    each lane's v16h is one contiguous 32B chunk -> GEMM loads are b128-wide.
//  - GEMM: 32x64 wave tile (2 A-frags x 4 B-frags -> 8 WMMAs per k-chunk),
//    k-loop fully unrolled via template so loads pipeline ahead of WMMAs.
//  - v_wmma_f32_16x16x32_f16, f16 activations, f32 accumulation.
//  - BN: atomic channel sums -> bnparams (scale/shift) -> fused pack kernels.
//  - Layer-3 output kept raw f32; BN affine folded into consumers.
// ---------------------------------------------------------------------------

typedef __attribute__((ext_vector_type(16))) _Float16 v16h;
typedef __attribute__((ext_vector_type(8)))  float    v8f;

#define B_     4
#define N_     16384
#define M_     1024
#define K_     32
#define CIN_   64
#define COLS_  (B_*M_*K_)   // 131072 (batch folded into columns)
#define COLSH_ (B_*M_)      // 4096

__device__ __forceinline__ int kmapA(int h, int hi) {
    return hi*16 + h;
}

// ------------------------------ KNN + grouping -----------------------------
__global__ __launch_bounds__(32)
void knn_group_kernel(const float* __restrict__ xyz, const float* __restrict__ feat,
                      const int* __restrict__ sidx,
                      float* __restrict__ grouped,      // (B,69,M,K) in d_out
                      float* __restrict__ knn_xyz,      // (COLS_,3)
                      int* __restrict__ knn_idx,
                      float* __restrict__ density)      // (B*M)
{
    extern __shared__ float sdist[];                    // N_ floats = 64KB
    const int bm = blockIdx.x, lane = threadIdx.x;
    const int b = bm >> 10, m = bm & (M_ - 1);
    const int sp = sidx[m];
    const float sx = xyz[(b*N_+sp)*3+0];
    const float sy = xyz[(b*N_+sp)*3+1];
    const float sz = xyz[(b*N_+sp)*3+2];

    for (int i = lane; i < N_; i += 32) {
        float dx = xyz[(b*N_+i)*3+0]-sx;
        float dy = xyz[(b*N_+i)*3+1]-sy;
        float dz = xyz[(b*N_+i)*3+2]-sz;
        sdist[i] = dx*dx + dy*dy + dz*dz;
    }
    __syncthreads();

    int mysel = 0;
    for (int r = 0; r < K_; ++r) {
        float bv = 3.0e38f; int bi = 0x7fffffff;
        for (int i = lane; i < N_; i += 32) {
            float v = sdist[i];
            if (v < bv) { bv = v; bi = i; }
        }
        for (int off = 16; off; off >>= 1) {             // wave32 argmin, tie->low idx
            float ov = __shfl_xor(bv, off);
            int   oi = __shfl_xor(bi, off);
            if (ov < bv || (ov == bv && oi < bi)) { bv = ov; bi = oi; }
        }
        if (lane == r) mysel = bi;
        if (lane == 0) sdist[bi] = 3.4e38f;
        __syncthreads();
    }

    // lane k owns neighbor k
    const int sel = mysel;
    const int col = bm*K_ + lane;
    const float kx = xyz[(b*N_+sel)*3+0];
    const float ky = xyz[(b*N_+sel)*3+1];
    const float kz = xyz[(b*N_+sel)*3+2];
    const float rx = kx-sx, ry = ky-sy, rz = kz-sz;
    const float sq = rx*rx + ry*ry + rz*rz;
    const float dist = (sq == 0.f) ? 0.f : sqrtf(sq);
    float dsum = dist;
    for (int off = 16; off; off >>= 1) dsum += __shfl_xor(dsum, off);
    const float dens = 1.f / (dsum * (1.f/K_) + 1e-6f);

    knn_idx[col] = sel;
    knn_xyz[col*3+0] = kx; knn_xyz[col*3+1] = ky; knn_xyz[col*3+2] = kz;
    if (lane == 0) density[bm] = dens;

    auto wr = [&](int c, float v) {
        grouped[((size_t)(b*69 + c)*M_ + m)*K_ + lane] = v;
    };
    wr(0, rx); wr(1, ry); wr(2, rz); wr(3, dist); wr(4, dens);
    for (int j = 0; j < CIN_; ++j)
        wr(5 + j, feat[(size_t)(b*CIN_ + j)*N_ + sel]);
}

// --------------------------- fragment packers ------------------------------
// W (Cout x CinReal) f32 -> A-frag tiles [(rowTile*chunks + kc)*32 + lane].
__global__ __launch_bounds__(32)
void pack_w_kernel(const float* __restrict__ W, _Float16* __restrict__ Wf,
                   int CinReal, int chunks)
{
    const int rt = blockIdx.x, kc = blockIdx.y, l = threadIdx.x;
    const int hi = l >> 4, row = rt*16 + (l & 15);
    v16h a;
    #pragma unroll
    for (int h = 0; h < 16; ++h) {
        const int kg = kc*32 + kmapA(h, hi);
        a[h] = (kg < CinReal) ? (_Float16)W[(size_t)row*CinReal + kg] : (_Float16)0.f;
    }
    ((v16h*)Wf)[(size_t)(rt*chunks + kc)*32 + l] = a;
}

// grouped f32 (B,69,M,K) -> layer-1 B-frags (96 padded channels).
__global__ __launch_bounds__(256)
void pack_grouped_kernel(const float* __restrict__ grouped, _Float16* __restrict__ Xf)
{
    const int tid = threadIdx.x, l = tid & 31, w = tid >> 5;
    const int ct = blockIdx.x*8 + w;            // 16-col tile
    const int kc = blockIdx.y;                  // 32-channel chunk
    const int hi = l >> 4;
    const int colq = ct*16 + (l & 15);
    const int bm = colq >> 5, k = colq & 31;
    const int b = bm >> 10, m = bm & 1023;
    v16h v;
    #pragma unroll
    for (int h = 0; h < 16; ++h) {
        const int c = kc*32 + hi*16 + h;
        v[h] = (c < 69) ? (_Float16)grouped[((size_t)(b*69 + c)*M_ + m)*K_ + k]
                        : (_Float16)0.f;
    }
    ((v16h*)Xf)[((size_t)kc*(COLS_ >> 4) + ct)*32 + l] = v;
}

// per-channel BN scale/shift from raw sums.
__global__ __launch_bounds__(256)
void bnparams_kernel(const float* __restrict__ stats, const float* __restrict__ g,
                     const float* __restrict__ bb, float invCount, int C,
                     float* __restrict__ p)
{
    const int c = threadIdx.x;
    float s = 0.f, sh = 0.f;
    if (c < C) {
        const float mu  = stats[c]*invCount;
        const float var = stats[256+c]*invCount - mu*mu;
        s  = g[c]*rsqrtf(var + 1e-5f);
        sh = bb[c] - mu*s;
    }
    p[c] = s; p[256+c] = sh;
}

// relu(scale*Y+shift) -> f16 B-frags. grid=(cols/128, C/32), block 256.
__global__ __launch_bounds__(256)
void bnpack_kernel(const float* __restrict__ Y, const float* __restrict__ p,
                   _Float16* __restrict__ Xf, int cols)
{
    const int tid = threadIdx.x, l = tid & 31, w = tid >> 5;
    const int ct = blockIdx.x*8 + w;
    const int kc = blockIdx.y;
    const int hi = l >> 4;
    const int colq = ct*16 + (l & 15);
    v16h v;
    #pragma unroll
    for (int h = 0; h < 16; ++h) {
        const int c = kc*32 + hi*16 + h;
        const float x = p[c]*Y[(size_t)c*cols + colq] + p[256+c];
        v[h] = (_Float16)fmaxf(x, 0.f);
    }
    ((v16h*)Xf)[((size_t)kc*(cols >> 4) + ct)*32 + l] = v;
}

// af f32 (B,256,M) -> head B-frags (cols = B*M).
__global__ __launch_bounds__(256)
void pack_head_kernel(const float* __restrict__ af, _Float16* __restrict__ Xf)
{
    const int tid = threadIdx.x, l = tid & 31, w = tid >> 5;
    const int ct = blockIdx.x*8 + w;
    const int kc = blockIdx.y;
    const int hi = l >> 4;
    const int colq = ct*16 + (l & 15);
    const int b = colq >> 10, m = colq & 1023;
    v16h v;
    #pragma unroll
    for (int h = 0; h < 16; ++h) {
        const int c = kc*32 + hi*16 + h;
        v[h] = (_Float16)af[(size_t)(b*256 + c)*M_ + m];
    }
    ((v16h*)Xf)[((size_t)kc*(COLSH_ >> 4) + ct)*32 + l] = v;
}

// ------------------------------ WMMA GEMM ----------------------------------
// Y[Cout x cols] = A-frags x B-frags.
// grid = (cols/256, Cout/64); block 256 (8 waves); wave tile = 32x64.
// Fully unrolled k-loop (CHUNKS template) so all 6 fragment loads of a chunk
// issue ahead of its 8 WMMAs and pipeline across chunks.
template<int CHUNKS>
__global__ __launch_bounds__(256)
void wmma_gemm_frag(const _Float16* __restrict__ Wf, const _Float16* __restrict__ Xf,
                    float* __restrict__ Y, int cols)
{
    const int tid  = threadIdx.x;
    const int lane = tid & 31, w = tid >> 5;
    const int rowTile0 = blockIdx.y*4 + (w >> 2)*2;     // two 16-row tiles
    const int colBase  = blockIdx.x*256 + (w & 3)*64;
    const int ct0 = colBase >> 4;
    const int tpc = cols >> 4;
    const v16h* Wt = (const v16h*)Wf;
    const v16h* Xt = (const v16h*)Xf;

    v8f acc[2][4];
    #pragma unroll
    for (int r = 0; r < 2; ++r)
        #pragma unroll
        for (int s = 0; s < 4; ++s) acc[r][s] = v8f{};

    #pragma unroll
    for (int kc = 0; kc < CHUNKS; ++kc) {
        if (kc + 1 < CHUNKS)
            __builtin_prefetch(&Xt[((size_t)(kc+1)*tpc + ct0)*32 + lane], 0, 1);
        const v16h a0 = Wt[(size_t)((rowTile0    )*CHUNKS + kc)*32 + lane];
        const v16h a1 = Wt[(size_t)((rowTile0 + 1)*CHUNKS + kc)*32 + lane];
        const v16h b0 = Xt[((size_t)kc*tpc + ct0 + 0)*32 + lane];
        const v16h b1 = Xt[((size_t)kc*tpc + ct0 + 1)*32 + lane];
        const v16h b2 = Xt[((size_t)kc*tpc + ct0 + 2)*32 + lane];
        const v16h b3 = Xt[((size_t)kc*tpc + ct0 + 3)*32 + lane];
        acc[0][0] = __builtin_amdgcn_wmma_f32_16x16x32_f16(false, a0, false, b0, (short)0, acc[0][0], false, false);
        acc[0][1] = __builtin_amdgcn_wmma_f32_16x16x32_f16(false, a0, false, b1, (short)0, acc[0][1], false, false);
        acc[0][2] = __builtin_amdgcn_wmma_f32_16x16x32_f16(false, a0, false, b2, (short)0, acc[0][2], false, false);
        acc[0][3] = __builtin_amdgcn_wmma_f32_16x16x32_f16(false, a0, false, b3, (short)0, acc[0][3], false, false);
        acc[1][0] = __builtin_amdgcn_wmma_f32_16x16x32_f16(false, a1, false, b0, (short)0, acc[1][0], false, false);
        acc[1][1] = __builtin_amdgcn_wmma_f32_16x16x32_f16(false, a1, false, b1, (short)0, acc[1][1], false, false);
        acc[1][2] = __builtin_amdgcn_wmma_f32_16x16x32_f16(false, a1, false, b2, (short)0, acc[1][2], false, false);
        acc[1][3] = __builtin_amdgcn_wmma_f32_16x16x32_f16(false, a1, false, b3, (short)0, acc[1][3], false, false);
    }

    const int hi  = lane >> 4;
    const int l15 = lane & 15;
    #pragma unroll
    for (int r = 0; r < 2; ++r) {
        const int rowB = (rowTile0 + r)*16 + hi*8;
        #pragma unroll
        for (int sub = 0; sub < 4; ++sub) {
            const int col = colBase + sub*16 + l15;
            #pragma unroll
            for (int e = 0; e < 8; ++e)
                Y[(size_t)(rowB + e)*cols + col] = acc[r][sub][e];
        }
    }
}

// ------------------------------ BN stats -----------------------------------
__global__ __launch_bounds__(256)
void zero_stats(float* __restrict__ s)
{
    int i = blockIdx.x*256 + threadIdx.x;
    if (i < 512) s[i] = 0.f;
}

__global__ __launch_bounds__(256)
void stats_kernel(const float* __restrict__ Y, float* __restrict__ stats,
                  int cols, int colsPerBlock)
{
    const int c = blockIdx.y, tid = threadIdx.x;
    const int base = blockIdx.x*colsPerBlock;
    float s = 0.f, s2 = 0.f;
    for (int i = tid; i < colsPerBlock; i += 256) {
        float v = Y[(size_t)c*cols + base + i];
        s += v; s2 += v*v;
    }
    __shared__ float r1[256], r2[256];
    r1[tid] = s; r2[tid] = s2; __syncthreads();
    for (int o = 128; o; o >>= 1) {
        if (tid < o) { r1[tid] += r1[tid+o]; r2[tid] += r2[tid+o]; }
        __syncthreads();
    }
    if (tid == 0) { atomicAdd(&stats[c], r1[0]); atomicAdd(&stats[256+c], r2[0]); }
}

// ------------------ Layer-3 consumers (BN folded on the fly) ---------------
__global__ __launch_bounds__(256)
void maxch_kernel(const float* __restrict__ Y, const float* __restrict__ p,
                  float* __restrict__ x1)
{
    __shared__ float ss[256], sh[256];
    const int tid = threadIdx.x;
    ss[tid] = p[tid]; sh[tid] = p[256+tid];
    __syncthreads();
    const int col = blockIdx.x*256 + tid;
    float mx = -3.4e38f;
    for (int c = 0; c < 256; ++c) {
        float h = fmaxf(ss[c]*Y[(size_t)c*COLS_ + col] + sh[c], 0.f);
        mx = fmaxf(mx, h);
    }
    x1[col] = mx;
}

__global__ __launch_bounds__(32)
void softmax_kp_kernel(const float* __restrict__ x1, const float* __restrict__ knn_xyz,
                       const float* __restrict__ density, float* __restrict__ aw,
                       float* __restrict__ kp_out, float* __restrict__ adj_out)
{
    const int bm = blockIdx.x, lane = threadIdx.x;
    const int col = bm*K_ + lane;
    const float v = x1[col];
    float mx = v;
    for (int o = 16; o; o >>= 1) mx = fmaxf(mx, __shfl_xor(mx, o));
    const float e = expf(v - mx);
    float ssum = e;
    for (int o = 16; o; o >>= 1) ssum += __shfl_xor(ssum, o);
    const float a = e / ssum;
    aw[col] = a;
    float kx = knn_xyz[col*3+0]*a, ky = knn_xyz[col*3+1]*a, kz = knn_xyz[col*3+2]*a;
    float asum = a;
    for (int o = 16; o; o >>= 1) {
        kx += __shfl_xor(kx, o); ky += __shfl_xor(ky, o);
        kz += __shfl_xor(kz, o); asum += __shfl_xor(asum, o);
    }
    if (lane == 0) {
        kp_out[bm*3+0] = kx; kp_out[bm*3+1] = ky; kp_out[bm*3+2] = kz;
        adj_out[bm] = density[bm]*asum;     // density_k constant over k
    }
}

// afm = h*aw -> d_out ; af = sum_k afm -> d_out. One wave per (bm, c).
__global__ __launch_bounds__(256)
void afm_af_kernel(const float* __restrict__ Y, const float* __restrict__ p,
                   const float* __restrict__ aw,
                   float* __restrict__ afm, float* __restrict__ af)
{
    const int tid = threadIdx.x, lane = tid & 31, w = tid >> 5;
    const int gid = blockIdx.x*8 + w;
    const int c = gid & 255, bm = gid >> 8;
    const float s = p[c], sh = p[256+c];
    const int col = bm*K_ + lane;
    const float h = fmaxf(s*Y[(size_t)c*COLS_ + col] + sh, 0.f);
    const float v = h*aw[col];
    const int b = bm >> 10, m = bm & 1023;
    afm[(size_t)((b*256 + c)*M_ + m)*K_ + lane] = v;
    float sum = v;
    for (int o = 16; o; o >>= 1) sum += __shfl_xor(sum, o);
    if (lane == 0) af[(size_t)(b*256 + c)*M_ + m] = sum;
}

// relu(bn(Yh2)) dot Wm3 + bm3 -> softplus + 0.001.
__global__ __launch_bounds__(256)
void head_final_kernel(const float* __restrict__ Yh, const float* __restrict__ p,
                       const float* __restrict__ Wm3, const float* __restrict__ bm3,
                       float* __restrict__ sig_out)
{
    __shared__ float ss[256], sh[256];
    const int tid = threadIdx.x;
    ss[tid] = p[tid]; sh[tid] = p[256+tid];
    __syncthreads();
    const int col = blockIdx.x*256 + tid;
    float acc = 0.f;
    for (int c = 0; c < 256; ++c) {
        float h = fmaxf(ss[c]*Yh[(size_t)c*COLSH_ + col] + sh[c], 0.f);
        acc += h*Wm3[c];
    }
    acc += bm3[0];
    const float sp = (acc > 20.f) ? acc : log1pf(expf(acc));
    sig_out[col] = sp + 0.001f;
}

// ------------------------------ launch -------------------------------------
extern "C" void kernel_launch(void* const* d_in, const int* in_sizes, int n_in,
                              void* d_out, int out_size, void* d_ws, size_t ws_size,
                              hipStream_t stream)
{
    const float* xyz   = (const float*)d_in[0];
    const float* feat  = (const float*)d_in[1];
    const int*   sidx  = (const int*)  d_in[2];
    const float* W1 = (const float*)d_in[3],  *g1 = (const float*)d_in[4],  *b1 = (const float*)d_in[5];
    const float* W2 = (const float*)d_in[6],  *g2 = (const float*)d_in[7],  *b2 = (const float*)d_in[8];
    const float* W3 = (const float*)d_in[9],  *g3 = (const float*)d_in[10], *b3 = (const float*)d_in[11];
    const float* Wm1 = (const float*)d_in[12];                        // bm1 d_in[13]: cancels in BN
    const float* gm1 = (const float*)d_in[14], *bem1 = (const float*)d_in[15];
    const float* Wm2 = (const float*)d_in[16];                        // bm2 d_in[17]: cancels in BN
    const float* gm2 = (const float*)d_in[18], *bem2 = (const float*)d_in[19];
    const float* Wm3 = (const float*)d_in[20], *bm3 = (const float*)d_in[21];

    float* out = (float*)d_out;
    float* out_kp      = out;                // (B,M,3)   12288
    float* out_sig     = out + 12288;        // (B,M)      4096
    float* out_af      = out + 16384;        // (B,256,M)  1048576
    float* out_grouped = out + 1064960;      // (B,69,M,K) 9043968
    float* out_afm     = out + 10108928;     // (B,256,M,K)33554432
    float* out_adj     = out + 43663360;     // (B,M)      4096

    char* ws = (char*)d_ws;
    int*      knn_idx = (int*)      (ws + 0);
    float*    knn_xyz = (float*)    (ws + 524288);
    float*    density = (float*)    (ws + 2097152);
    float*    x1      = (float*)    (ws + 2113536);
    float*    aw      = (float*)    (ws + 2637824);
    float*    stats   = (float*)    (ws + 3162112);   // 512 floats
    float*    params  = (float*)    (ws + 3164160);   // 512 floats (scale/shift)
    _Float16* Xf16    = (_Float16*) (ws + 3166208);   // <=256 x COLS_ f16 frags
    float*    Yf32    = (float*)    (ws + 70275072);  // 256 x COLS_ f32
    _Float16* afh     = (_Float16*) (ws + 204492800); // 256 x 4096 f16 frags
    _Float16* Xh      = (_Float16*) (ws + 206589952); // 256 x 4096 f16 frags
    float*    Yh      = (float*)    (ws + 208687104); // 256 x 4096 f32
    _Float16* Wf      = (_Float16*) (ws + 212881408); // <=128KB A-frags

    // 1) KNN + grouping
    knn_group_kernel<<<B_*M_, 32, N_*sizeof(float), stream>>>(
        xyz, feat, sidx, out_grouped, knn_xyz, knn_idx, density);
    pack_grouped_kernel<<<dim3(COLS_/128, 3), 256, 0, stream>>>(out_grouped, Xf16);

    // 2) layer1: 64 x 69 (padded 96)
    pack_w_kernel<<<dim3(4, 3), 32, 0, stream>>>(W1, Wf, 69, 3);
    wmma_gemm_frag<3><<<dim3(COLS_/256, 1), 256, 0, stream>>>(Wf, Xf16, Yf32, COLS_);
    zero_stats<<<2, 256, 0, stream>>>(stats);
    stats_kernel<<<dim3(16, 64), 256, 0, stream>>>(Yf32, stats, COLS_, COLS_/16);
    bnparams_kernel<<<1, 256, 0, stream>>>(stats, g1, b1, 1.f/COLS_, 64, params);
    bnpack_kernel<<<dim3(COLS_/128, 2), 256, 0, stream>>>(Yf32, params, Xf16, COLS_);

    // 3) layer2: 128 x 64
    pack_w_kernel<<<dim3(8, 2), 32, 0, stream>>>(W2, Wf, 64, 2);
    wmma_gemm_frag<2><<<dim3(COLS_/256, 2), 256, 0, stream>>>(Wf, Xf16, Yf32, COLS_);
    zero_stats<<<2, 256, 0, stream>>>(stats);
    stats_kernel<<<dim3(16, 128), 256, 0, stream>>>(Yf32, stats, COLS_, COLS_/16);
    bnparams_kernel<<<1, 256, 0, stream>>>(stats, g2, b2, 1.f/COLS_, 128, params);
    bnpack_kernel<<<dim3(COLS_/128, 4), 256, 0, stream>>>(Yf32, params, Xf16, COLS_);

    // 4) layer3: 256 x 128 (keep raw f32; fold BN into consumers)
    pack_w_kernel<<<dim3(16, 4), 32, 0, stream>>>(W3, Wf, 128, 4);
    wmma_gemm_frag<4><<<dim3(COLS_/256, 4), 256, 0, stream>>>(Wf, Xf16, Yf32, COLS_);
    zero_stats<<<2, 256, 0, stream>>>(stats);
    stats_kernel<<<dim3(16, 256), 256, 0, stream>>>(Yf32, stats, COLS_, COLS_/16);
    bnparams_kernel<<<1, 256, 0, stream>>>(stats, g3, b3, 1.f/COLS_, 256, params);

    // 5) channel-max -> softmax over K -> keypoints/adj_density -> afm/af
    maxch_kernel<<<COLS_/256, 256, 0, stream>>>(Yf32, params, x1);
    softmax_kp_kernel<<<B_*M_, 32, 0, stream>>>(x1, knn_xyz, density, aw, out_kp, out_adj);
    afm_af_kernel<<<(B_*M_*256)/8, 256, 0, stream>>>(Yf32, params, aw, out_afm, out_af);

    // 6) MLP head (COLSH_ = 4096 columns)
    pack_head_kernel<<<dim3(COLSH_/128, 8), 256, 0, stream>>>(out_af, afh);
    pack_w_kernel<<<dim3(16, 8), 32, 0, stream>>>(Wm1, Wf, 256, 8);
    wmma_gemm_frag<8><<<dim3(COLSH_/256, 4), 256, 0, stream>>>(Wf, afh, Yh, COLSH_);
    zero_stats<<<2, 256, 0, stream>>>(stats);
    stats_kernel<<<dim3(1, 256), 256, 0, stream>>>(Yh, stats, COLSH_, COLSH_);
    bnparams_kernel<<<1, 256, 0, stream>>>(stats, gm1, bem1, 1.f/COLSH_, 256, params);
    bnpack_kernel<<<dim3(COLSH_/128, 8), 256, 0, stream>>>(Yh, params, Xh, COLSH_);

    pack_w_kernel<<<dim3(16, 8), 32, 0, stream>>>(Wm2, Wf, 256, 8);
    wmma_gemm_frag<8><<<dim3(COLSH_/256, 4), 256, 0, stream>>>(Wf, Xh, Yh, COLSH_);
    zero_stats<<<2, 256, 0, stream>>>(stats);
    stats_kernel<<<dim3(1, 256), 256, 0, stream>>>(Yh, stats, COLSH_, COLSH_);
    bnparams_kernel<<<1, 256, 0, stream>>>(stats, gm2, bem2, 1.f/COLSH_, 256, params);
    head_final_kernel<<<COLSH_/256, 256, 0, stream>>>(Yh, params, Wm3, bm3, out_sig);
}
